// FGSBIR_GAT_893353198181
// MI455X (gfx1250) — compile-verified, matrix operand order checked by reading, not run on hardware
//
#include <hip/hip_runtime.h>

// ---------------------------------------------------------------------------
// FGSBIR GAT forward for gfx1250 (MI455X), wave32 + WMMA bf16 + async-LDS.
// B=8, N=1024, Dlab=256, Dvis=512, H=8, F=64, Dgat=512, Demb=512.
// ---------------------------------------------------------------------------

#define B_   8
#define N_   1024
#define DLAB 256
#define DVIS 512
#define H_   8
#define F_   64
#define DG   512   // H*F
#define DEMB 512
#define ALPHA 0.2f

#define ASYNC_STAGE 1   // use global_load_async_to_lds_b128 for edl staging

typedef __attribute__((ext_vector_type(16))) __bf16 v16bf;
typedef __attribute__((ext_vector_type(8)))  __bf16 v8bf;
typedef __attribute__((ext_vector_type(8)))  float  v8f;

// ---------------------------------------------------------------------------
// Kernel 1: Wh_lab (1024x512) = labels(1024x256) @ W[0:256,:]  via bf16 WMMA
// one wave per 16x16 output tile, K loop in chunks of 32.
// ---------------------------------------------------------------------------
__global__ void k_gemm_whlab(const float* __restrict__ lab,
                             const float* __restrict__ W,
                             float* __restrict__ whlab) {
  int lane = threadIdx.x & 31;
  int wid  = (blockIdx.x * blockDim.x + threadIdx.x) >> 5;   // 0..2047
  int i0 = (wid >> 5) * 16;     // 64 row tiles
  int c0 = (wid & 31) * 16;     // 32 col tiles
  int r    = lane & 15;
  int koff = (lane < 16) ? 0 : 8;

  v8f acc = {};
  for (int k0 = 0; k0 < DLAB; k0 += 32) {
    const float* ap = lab + (i0 + r) * DLAB + k0 + koff;
    v16bf A;
#pragma unroll
    for (int e = 0; e < 8; ++e) {
      A[e]     = (__bf16)ap[e];        // k = koff + e
      A[8 + e] = (__bf16)ap[16 + e];   // k = koff + 16 + e
    }
    v16bf Bf;
#pragma unroll
    for (int e = 0; e < 16; ++e) {
      int ke = ((e < 8) ? e : (e + 8)) + koff;
      Bf[e] = (__bf16)W[(size_t)(k0 + ke) * DG + c0 + r];
    }
    acc = __builtin_amdgcn_wmma_f32_16x16x32_bf16(false, A, false, Bf,
                                                  (short)0, acc, false, false);
  }
  int mbase = (lane < 16) ? 0 : 8;
#pragma unroll
  for (int v = 0; v < 8; ++v)
    whlab[(size_t)(i0 + mbase + v) * DG + c0 + r] = acc[v];
}

// ---------------------------------------------------------------------------
// Kernel 2: Wh_vis (8x512) = visual(8x512) @ W[256:768,:]
// ---------------------------------------------------------------------------
__global__ void k_whvis(const float* __restrict__ vis,
                        const float* __restrict__ W,
                        float* __restrict__ whvis) {
  int tid = blockIdx.x * blockDim.x + threadIdx.x;   // 4096
  int b = tid >> 9, c = tid & 511;
  float acc = 0.f;
  for (int i = 0; i < DVIS; ++i)
    acc += vis[b * DVIS + i] * W[(size_t)(DLAB + i) * DG + c];
  whvis[b * DG + c] = acc;
}

// ---------------------------------------------------------------------------
// Kernel 3: per-node logits  esl/edl[n][h] = Wh_lab[n,h,:]·a_{src,dst}[h]
// ---------------------------------------------------------------------------
__global__ void k_srcdst(const float* __restrict__ whlab,
                         const float* __restrict__ a_src,
                         const float* __restrict__ a_dst,
                         float* __restrict__ esl, float* __restrict__ edl) {
  int tid = blockIdx.x * blockDim.x + threadIdx.x;   // 8192
  int n = tid >> 3, h = tid & 7;
  float s1 = 0.f, s2 = 0.f;
  for (int f = 0; f < F_; ++f) {
    float w = whlab[(size_t)n * DG + h * F_ + f];
    s1 += w * a_src[h * F_ + f];
    s2 += w * a_dst[h * F_ + f];
  }
  esl[n * H_ + h] = s1;
  edl[n * H_ + h] = s2;
}

__global__ void k_srcdst_vis(const float* __restrict__ whvis,
                             const float* __restrict__ a_src,
                             const float* __restrict__ a_dst,
                             float* __restrict__ esv, float* __restrict__ edv) {
  int tid = threadIdx.x;                              // 64
  int b = tid >> 3, h = tid & 7;
  float s1 = 0.f, s2 = 0.f;
  for (int f = 0; f < F_; ++f) {
    float w = whvis[b * DG + h * F_ + f];
    s1 += w * a_src[h * F_ + f];
    s2 += w * a_dst[h * F_ + f];
  }
  esv[b * H_ + h] = s1;
  edv[b * H_ + h] = s2;
}

// ---------------------------------------------------------------------------
// Kernel 3b: per-head max of edl (softmax upper bound; lrelu is monotone)
// ---------------------------------------------------------------------------
__global__ void k_dmax(const float* __restrict__ edl, float* __restrict__ medl) {
  int h = threadIdx.x;                                // 8
  float m = -1e30f;
  for (int j = 0; j < N_; ++j) m = fmaxf(m, edl[j * H_ + h]);
  medl[h] = m;
}

// ---------------------------------------------------------------------------
// Kernel 4: adjacency -> bitmask (1024 x 32 words)
// ---------------------------------------------------------------------------
__global__ void k_bits(const float* __restrict__ adj, unsigned* __restrict__ bits) {
  int tid = blockIdx.x * blockDim.x + threadIdx.x;   // 32768
  int i = tid >> 5, w = tid & 31;
  unsigned m = 0u;
  for (int t = 0; t < 32; ++t)
    if (adj[(size_t)i * N_ + w * 32 + t] > 0.f) m |= (1u << t);
  bits[i * 32 + w] = m;
}

// ---------------------------------------------------------------------------
// Kernel 5: WhbT[b][h][f][j] = bf16(Wh_lab[j][h*64+f] + Wh_vis[b][h*64+f])
// j innermost so WMMA B-fragments are contiguous 16B loads.
// ---------------------------------------------------------------------------
__global__ void k_whbT(const float* __restrict__ whlab,
                       const float* __restrict__ whvis,
                       __bf16* __restrict__ whbT) {
  unsigned idx = blockIdx.x * blockDim.x + threadIdx.x;  // 4M
  unsigned j = idx & 1023u, f = (idx >> 10) & 63u;
  unsigned h = (idx >> 16) & 7u, b = idx >> 19;
  float v = whlab[(size_t)j * DG + h * F_ + f] + whvis[b * DG + h * F_ + f];
  whbT[idx] = (__bf16)v;
}

// ---------------------------------------------------------------------------
// Kernel 6 (main): fused masked softmax + WMMA aggregation + ELU + pool score.
// grid: (64 i-tiles, 8 batches), 256 threads = 8 waves, wave w == head h.
// ---------------------------------------------------------------------------
__global__ void __launch_bounds__(256)
k_agg(const __bf16* __restrict__ whbT,
      const unsigned* __restrict__ adj_bits,
      const float* __restrict__ esl, const float* __restrict__ edl,
      const float* __restrict__ esv, const float* __restrict__ edv,
      const float* __restrict__ medl,
      const float* __restrict__ pool_q,
      __bf16* __restrict__ xbf, float* __restrict__ scores) {
  __shared__ float    s_edl[H_ * N_];    // raw edl copy, [j][h] layout, 32KB
  __shared__ unsigned s_mask[16 * 32];   // 16 rows of adjacency bits
  __shared__ float    s_es[16 * H_];     // per-row source logits
  __shared__ float    s_score[16];       // pooling score accumulator

  const int tid = threadIdx.x;
  const int lane = tid & 31;
  const int h = tid >> 5;
  const int i0 = blockIdx.x * 16;
  const int b = blockIdx.y;

#if ASYNC_STAGE
  {
    // DMA edl (32KB) into LDS: per-lane 16B transfers tracked by ASYNCcnt.
    unsigned ldsbase = (unsigned)(uintptr_t)(&s_edl[0]);
    for (int idx = tid; idx < (H_ * N_) / 4; idx += 256) {
      unsigned loff = ldsbase + idx * 16u;
      const float* g = edl + idx * 4;
      asm volatile("global_load_async_to_lds_b128 %0, %1, off"
                   :: "v"(loff), "v"(g) : "memory");
    }
  }
#else
  for (int idx = tid; idx < H_ * N_; idx += 256) s_edl[idx] = edl[idx];
#endif
  for (int idx = tid; idx < 512; idx += 256)
    s_mask[idx] = adj_bits[i0 * 32 + idx];
  if (tid < 128) {
    int rr = tid >> 3, hh = tid & 7;
    s_es[tid] = esl[(i0 + rr) * H_ + hh] + esv[b * H_ + hh];
  }
  if (tid < 16) s_score[tid] = 0.f;
#if ASYNC_STAGE
  asm volatile("s_wait_asynccnt 0" ::: "memory");
#endif
  __syncthreads();

  const int r = lane & 15;            // my attention row within the tile
  const int koff = (lane < 16) ? 0 : 8;

  // stable-softmax upper bound: lrelu monotone => lrelu(s + max_j d_j) bounds
  // every masked logit; exact max not needed (shift-invariance of softmax).
  const float sp = s_es[r * H_ + h] + edv[b * H_ + h];
  float mb = sp + medl[h];
  mb = (mb > 0.f) ? mb : ALPHA * mb;

  // ---- probabilities -> A frag, V -> B frag, WMMA accumulate ----
  v8f acc[4] = {};
  float rowsum = 0.f;
  const size_t vbase = ((size_t)(b * H_ + h)) * F_ * N_;

  for (int j0 = 0; j0 < N_; j0 += 32) {
    v16bf A;
#pragma unroll
    for (int e = 0; e < 16; ++e) {
      int k = ((e < 8) ? e : (e + 8)) + koff;
      int j = j0 + k;
      float t = sp + s_edl[j * H_ + h];
      t = (t > 0.f) ? t : ALPHA * t;
      bool ok = (s_mask[r * 32 + (j >> 5)] >> (j & 31)) & 1u;
      float p = ok ? __expf(t - mb) : 0.f;
      rowsum += p;
      A[e] = (__bf16)p;
    }
#pragma unroll
    for (int t4 = 0; t4 < 4; ++t4) {
      int f = t4 * 16 + r;
      const __bf16* bp = whbT + vbase + (size_t)f * N_ + j0 + koff;
      if (t4 == 0) __builtin_prefetch(bp + 32, 0, 1);   // global_prefetch_b8
      v8bf lo = *(const v8bf*)bp;
      v8bf hi = *(const v8bf*)(bp + 16);
      v16bf Bf;
#pragma unroll
      for (int e = 0; e < 8; ++e) { Bf[e] = lo[e]; Bf[8 + e] = hi[e]; }
      // A is identical across the 4 WMMAs: set reuse_a on repeats (ISA 7.12).
      // reuse flag must be a literal ICE -> branch on unroll-constant t4.
      if (t4 == 0)
        acc[t4] = __builtin_amdgcn_wmma_f32_16x16x32_bf16(
            false, A, false, Bf, (short)0, acc[t4], false, false);
      else
        acc[t4] = __builtin_amdgcn_wmma_f32_16x16x32_bf16(
            false, A, false, Bf, (short)0, acc[t4], true, false);
    }
  }

  // ---- normalize, ELU, store x (bf16), accumulate pooling score ----
  rowsum += __shfl_xor(rowsum, 16, 32);   // lane L holds sum of row L&15
  const int mbase = (lane < 16) ? 0 : 8;
  float inv[8];
#pragma unroll
  for (int v = 0; v < 8; ++v)
    inv[v] = 1.f / __shfl(rowsum, mbase + v, 32);

  float spart[8] = {0.f, 0.f, 0.f, 0.f, 0.f, 0.f, 0.f, 0.f};
#pragma unroll
  for (int t4 = 0; t4 < 4; ++t4) {
    int c = h * F_ + t4 * 16 + r;       // output feature column
    float q = pool_q[c];
#pragma unroll
    for (int v = 0; v < 8; ++v) {
      float o = acc[t4][v] * inv[v];
      o = (o > 0.f) ? o : (__expf(o) - 1.f);   // ELU
      xbf[((size_t)b * N_ + i0 + mbase + v) * DG + c] = (__bf16)o;
      spart[v] += o * q;
    }
  }
#pragma unroll
  for (int v = 0; v < 8; ++v) {
    for (int o = 1; o < 16; o <<= 1)
      spart[v] += __shfl_xor(spart[v], o, 32);   // reduce within half-wave
    if (r == 0) atomicAdd(&s_score[mbase + v], spart[v]);
  }
  __syncthreads();
  if (tid < 16) scores[b * N_ + i0 + tid] = s_score[tid];
}

// ---------------------------------------------------------------------------
// Kernel 7: per-batch softmax pooling + FC. grid = 8 blocks x 256 threads.
// ---------------------------------------------------------------------------
__global__ void k_pool(const float* __restrict__ scores,
                       const __bf16* __restrict__ xbf,
                       const float* __restrict__ fcW,
                       const float* __restrict__ fcb,
                       float* __restrict__ out) {
  __shared__ float red[256];
  __shared__ float w[N_];
  __shared__ float pooled[DG];
  const int tid = threadIdx.x, b = blockIdx.x;
  const float* sc = scores + b * N_;

  float m = -1e30f;
  for (int k = 0; k < 4; ++k) m = fmaxf(m, sc[tid + k * 256]);
  red[tid] = m; __syncthreads();
  for (int s = 128; s > 0; s >>= 1) {
    if (tid < s) red[tid] = fmaxf(red[tid], red[tid + s]);
    __syncthreads();
  }
  m = red[0]; __syncthreads();

  float sum = 0.f;
  for (int k = 0; k < 4; ++k) {
    float e = __expf(sc[tid + k * 256] - m);
    w[tid + k * 256] = e;
    sum += e;
  }
  red[tid] = sum; __syncthreads();
  for (int s = 128; s > 0; s >>= 1) {
    if (tid < s) red[tid] += red[tid + s];
    __syncthreads();
  }
  float invs = 1.f / red[0]; __syncthreads();

  for (int k = 0; k < 2; ++k) {
    int c = tid + k * 256;
    float acc = 0.f;
    const __bf16* xc = xbf + (size_t)b * N_ * DG + c;
    for (int n = 0; n < N_; ++n) acc += w[n] * (float)xc[(size_t)n * DG];
    pooled[c] = acc * invs;
  }
  __syncthreads();

  for (int k = 0; k < 2; ++k) {
    int e = tid + k * 256;
    float acc = fcb[e];
    for (int c = 0; c < DG; ++c) acc += pooled[c] * fcW[(size_t)c * DEMB + e];
    out[b * DEMB + e] = acc;
  }
}

// ---------------------------------------------------------------------------
extern "C" void kernel_launch(void* const* d_in, const int* in_sizes, int n_in,
                              void* d_out, int out_size, void* d_ws, size_t ws_size,
                              hipStream_t stream) {
  const float* visual = (const float*)d_in[0];
  const float* labels = (const float*)d_in[1];
  const float* adj    = (const float*)d_in[2];
  const float* W      = (const float*)d_in[3];
  const float* a_src  = (const float*)d_in[4];
  const float* a_dst  = (const float*)d_in[5];
  const float* pool_q = (const float*)d_in[6];
  const float* fc_W   = (const float*)d_in[7];
  const float* fc_b   = (const float*)d_in[8];
  float* out = (float*)d_out;

  char* w = (char*)d_ws;
  float* Wh_lab = (float*)w;            w += (size_t)N_ * DG * 4;       // 2 MB
  float* Wh_vis = (float*)w;            w += (size_t)B_ * DG * 4;       // 16 KB
  float* esl    = (float*)w;            w += (size_t)N_ * H_ * 4;       // 32 KB
  float* edl    = (float*)w;            w += (size_t)N_ * H_ * 4;
  float* esv    = (float*)w;            w += 4096;
  float* edv    = (float*)w;            w += 4096;
  float* medl   = (float*)w;            w += 4096;
  unsigned* adj_bits = (unsigned*)w;    w += (size_t)N_ * 32 * 4;       // 128 KB
  __bf16* whbT  = (__bf16*)w;           w += (size_t)B_ * H_ * F_ * N_ * 2; // 8 MB
  __bf16* xbf   = (__bf16*)w;           w += (size_t)B_ * N_ * DG * 2;      // 8 MB
  float* scores = (float*)w;            w += (size_t)B_ * N_ * 4;       // 32 KB

  k_gemm_whlab<<<256, 256, 0, stream>>>(labels, W, Wh_lab);
  k_whvis<<<16, 256, 0, stream>>>(visual, W, Wh_vis);
  k_srcdst<<<32, 256, 0, stream>>>(Wh_lab, a_src, a_dst, esl, edl);
  k_srcdst_vis<<<1, 64, 0, stream>>>(Wh_vis, a_src, a_dst, esv, edv);
  k_dmax<<<1, 8, 0, stream>>>(edl, medl);
  k_bits<<<128, 256, 0, stream>>>(adj, adj_bits);
  k_whbT<<<16384, 256, 0, stream>>>(Wh_lab, Wh_vis, whbT);
  k_agg<<<dim3(64, 8, 1), 256, 0, stream>>>(whbT, adj_bits, esl, edl, esv, edv,
                                            medl, pool_q, xbf, scores);
  k_pool<<<8, 256, 0, stream>>>(scores, xbf, fc_W, fc_b, out);
}
